// Net_38989713113414
// MI455X (gfx1250) — matrix operand order repaired and checked by reading, not compiled, not run on hardware
//
#include <hip/hip_runtime.h>
#include <hip/hip_bf16.h>

// Problem sizes (match reference)
#define Nn   40960
#define Ee   327680
#define N1n  20480
#define E2n  163840
#define N2n  10240
#define Bb   512
#define NCLSn 10

typedef __attribute__((ext_vector_type(16))) _Float16 v16h;
typedef __attribute__((ext_vector_type(8)))  float    v8f;

#define ENC_NEGINF 0x007FFFFFu

__device__ __forceinline__ unsigned encf(float f) {
  unsigned u = __float_as_uint(f);
  return (u & 0x80000000u) ? ~u : (u | 0x80000000u);
}
__device__ __forceinline__ float decf(unsigned u) {
  unsigned b = (u & 0x80000000u) ? (u & 0x7FFFFFFFu) : ~u;
  return __uint_as_float(b);
}
__device__ __forceinline__ float eluf(float v) { return v > 0.0f ? v : expm1f(v); }

// ---------------------------------------------------------------- utilities
__global__ void fill_u32(unsigned* p, unsigned v, int n) {
  int i = blockIdx.x * blockDim.x + threadIdx.x;
  if (i < n) p[i] = v;
}

// w2ext[26][2048] in f16 : rows 0..24 = nn2_w2, row 25 = nn2_b2 (bias folded as k=25, h1=1)
__global__ void prep_w2h(const float* __restrict__ w2, const float* __restrict__ b2,
                         _Float16* __restrict__ w2h) {
  int idx = blockIdx.x * blockDim.x + threadIdx.x;
  if (idx >= 26 * 2048) return;
  int k = idx / 2048, j = idx - k * 2048;
  float v = (k < 25) ? w2[k * 2048 + j] : b2[j];
  w2h[idx] = (_Float16)v;
}

// ---------------------------------------------------------------- conv1 (edge side)
__global__ __launch_bounds__(256) void conv1_edge(
    const float* __restrict__ x, const float* __restrict__ eattr, const int* __restrict__ ei,
    const float* __restrict__ w1, const float* __restrict__ b1,
    const float* __restrict__ w2, const float* __restrict__ b2,
    float* __restrict__ aggr1, float* __restrict__ cnt1) {
  __shared__ float w1s[50], b1s[25], w2s[25 * 32], b2s[32];
  int t = threadIdx.x;
  if (t < 50) w1s[t] = w1[t];
  if (t < 25) b1s[t] = b1[t];
  if (t < 32) b2s[t] = b2[t];
  for (int j = t; j < 800; j += 256) w2s[j] = w2[j];
  __syncthreads();
  int e = blockIdx.x * 256 + t;
  if (e >= Ee) return;
  float ea0 = eattr[e * 2 + 0], ea1 = eattr[e * 2 + 1];
  int src = ei[e], dst = ei[Ee + e];
  float h[25];
#pragma unroll
  for (int k = 0; k < 25; ++k)
    h[k] = fmaxf(ea0 * w1s[k] + ea1 * w1s[25 + k] + b1s[k], 0.0f);
  float xs = x[src];
#pragma unroll
  for (int o = 0; o < 32; ++o) {
    float acc = b2s[o];
#pragma unroll
    for (int k = 0; k < 25; ++k) acc += h[k] * w2s[k * 32 + o];
    atomicAdd(&aggr1[dst * 32 + o], xs * acc);
  }
  atomicAdd(&cnt1[dst], 1.0f);
}

// conv1 node side: x1 = elu(x*root1 + aggr/cnt + bias1)
__global__ void conv1_node(const float* __restrict__ x, const float* __restrict__ root1,
                           const float* __restrict__ bias1, const float* __restrict__ aggr1,
                           const float* __restrict__ cnt1, float* __restrict__ x1) {
  int idx = blockIdx.x * blockDim.x + threadIdx.x;
  if (idx >= Nn * 32) return;
  int n = idx >> 5, o = idx & 31;
  float c = fmaxf(cnt1[n], 1.0f);
  x1[idx] = eluf(x[n] * root1[o] + aggr1[idx] / c + bias1[o]);
}

// ---------------------------------------------------------------- graclus pool 1
__global__ void pool1(const float* __restrict__ x1, const float* __restrict__ pos,
                      const int* __restrict__ batch, const int* __restrict__ cluster1,
                      unsigned* __restrict__ x1p_enc, float* __restrict__ pos_sum,
                      float* __restrict__ pos_cnt, int* __restrict__ batch1) {
  int n = blockIdx.x * blockDim.x + threadIdx.x;
  if (n >= Nn) return;
  int c = cluster1[n];
#pragma unroll
  for (int o = 0; o < 32; ++o) atomicMax(&x1p_enc[c * 32 + o], encf(x1[n * 32 + o]));
  atomicAdd(&pos_sum[c * 2 + 0], pos[n * 2 + 0]);
  atomicAdd(&pos_sum[c * 2 + 1], pos[n * 2 + 1]);
  atomicAdd(&pos_cnt[c], 1.0f);
  atomicMax(&batch1[c], batch[n]);
}

__global__ void pos1_div(const float* __restrict__ pos_sum, const float* __restrict__ pos_cnt,
                         float* __restrict__ pos1) {
  int n = blockIdx.x * blockDim.x + threadIdx.x;
  if (n >= N1n) return;
  float c = fmaxf(pos_cnt[n], 1.0f);
  pos1[n * 2 + 0] = pos_sum[n * 2 + 0] / c;
  pos1[n * 2 + 1] = pos_sum[n * 2 + 1] / c;
}

__global__ void x1p_decode(const unsigned* __restrict__ enc, float* __restrict__ x1p) {
  int idx = blockIdx.x * blockDim.x + threadIdx.x;
  if (idx >= N1n * 32) return;
  unsigned u = enc[idx];
  x1p[idx] = (u == ENC_NEGINF) ? 0.0f : decf(u);
}

// global max |pos1[src]-pos1[dst]| over all E2 x 2 elements
__global__ void cart_max(const int* __restrict__ ei2, const float* __restrict__ pos1,
                         unsigned* __restrict__ m_enc) {
  int e = blockIdx.x * blockDim.x + threadIdx.x;
  if (e >= E2n) return;
  int s = ei2[e], d = ei2[E2n + e];
  float a0 = fabsf(pos1[s * 2 + 0] - pos1[d * 2 + 0]);
  float a1 = fabsf(pos1[s * 2 + 1] - pos1[d * 2 + 1]);
  atomicMax(m_enc, encf(fmaxf(a0, a1)));
}

// ---------------------------------------------------------------- conv2: fused WMMA kernel
// Per block: 128 edges. msg[e,o] = sum_{k<26} h1[e,k] * sum_i y[e,i]*w2ext[k, i*64+o]
// For each k: A (f16, 16x32) = h1[e,k]*y[e,i] built from k-invariant y registers,
// B (f16, 32x16) read as contiguous 32B runs from transposed LDS row, accumulate C (f32)
// with v_wmma_f32_16x16x32_f16.  4 waves x (2 M-tiles x 4 N-tiles).
__global__ __launch_bounds__(128) void conv2_edge_wmma(
    const int* __restrict__ ei2, const float* __restrict__ pos1,
    const unsigned* __restrict__ m_enc, const float* __restrict__ x1p,
    const float* __restrict__ w1, const float* __restrict__ b1,
    const _Float16* __restrict__ w2h,
    float* __restrict__ aggr2, float* __restrict__ cnt2) {
  __shared__ float h1s[128][26];
  __shared__ float ys[128][32];
  __shared__ int dsts[128];
  __shared__ __align__(32) _Float16 bks_t[64][32];  // transposed k-row: [o=64][i=32]

  const int t = threadIdx.x;
  const int e = blockIdx.x * 128 + t;  // E2n divisible by 128
  const int src = ei2[e];
  const int dst = ei2[E2n + e];
  dsts[t] = dst;

  float m = decf(*m_enc);
  float inv2m = (m > 0.0f) ? (0.5f / m) : 0.0f;
  float ea0 = (pos1[src * 2 + 0] - pos1[dst * 2 + 0]) * inv2m + 0.5f;
  float ea1 = (pos1[src * 2 + 1] - pos1[dst * 2 + 1]) * inv2m + 0.5f;
#pragma unroll
  for (int k = 0; k < 25; ++k)
    h1s[t][k] = fmaxf(ea0 * w1[k] + ea1 * w1[25 + k] + b1[k], 0.0f);
  h1s[t][25] = 1.0f;  // bias row of w2ext
#pragma unroll
  for (int i = 0; i < 32; ++i) ys[t][i] = x1p[src * 32 + i];
  atomicAdd(&cnt2[dst], 1.0f);
  __syncthreads();

  const int wave = t >> 5;        // 0..3 -> M-tiles {2w, 2w+1}
  const int lane = t & 31;
  const int lhalf = lane >> 4;    // 0 or 1
  const int l16 = lane & 15;

  // Hoist the k-invariant A-side y values into registers.
  // A 16x32 layout (matches B): elem p -> K = lhalf*16+p.
  float yv[2][16];
#pragma unroll
  for (int mt = 0; mt < 2; ++mt) {
    int r = (wave * 2 + mt) * 16 + l16;
#pragma unroll
    for (int p = 0; p < 16; ++p) {
      int kk = lhalf * 16 + p;
      yv[mt][p] = ys[r][kk];
    }
  }

  v8f acc[2][4] = {};

  for (int k = 0; k < 26; ++k) {
    __syncthreads();
    // Stage k-row of w2ext transposed: bks_t[o][i] = w2ext[k][i*64+o]
    for (int j = t; j < 2048; j += 128) {
      int i = j >> 6, o = j & 63;
      bks_t[o][i] = w2h[k * 2048 + j];
    }
    if (k + 1 < 26) __builtin_prefetch(&w2h[(k + 1) * 2048], 0, 3);
    __syncthreads();

    // A fragments: scale the cached y register values by h1[e,k], convert to f16.
    v16h a[2];
#pragma unroll
    for (int mt = 0; mt < 2; ++mt) {
      int r = (wave * 2 + mt) * 16 + l16;
      float hk = h1s[r][k];
#pragma unroll
      for (int p = 0; p < 16; ++p) a[mt][p] = (_Float16)(hk * yv[mt][p]);
    }
    // B fragments: elem j -> K = lhalf*16 + j, col N = nt*16 + l16.
    // Transposed layout makes the 16 elems contiguous (32B, 32B-aligned) -> ds_load_b128 x2.
    v16h bfr[4];
#pragma unroll
    for (int nt = 0; nt < 4; ++nt) {
      int col = nt * 16 + l16;
      bfr[nt] = *(const v16h*)(&bks_t[col][lhalf * 16]);
    }
#pragma unroll
    for (int mt = 0; mt < 2; ++mt)
#pragma unroll
      for (int nt = 0; nt < 4; ++nt)
        acc[mt][nt] = __builtin_amdgcn_wmma_f32_16x16x32_f16(
            false, a[mt], false, bfr[nt], (short)0, acc[mt][nt], false, false);
  }

  // Scatter C (layout: VGPR v -> M = lhalf*8 + v, N = l16) into segment-sum accumulator.
  __syncthreads();
#pragma unroll
  for (int mt = 0; mt < 2; ++mt) {
#pragma unroll
    for (int nt = 0; nt < 4; ++nt) {
      int col = nt * 16 + l16;
#pragma unroll
      for (int v = 0; v < 8; ++v) {
        int rlocal = (wave * 2 + mt) * 16 + lhalf * 8 + v;
        atomicAdd(&aggr2[dsts[rlocal] * 64 + col], acc[mt][nt][v]);
      }
    }
  }
}

// conv2 node side: x2 = elu(x1p@root2 + aggr2/cnt2 + bias2)
__global__ void conv2_node(const float* __restrict__ x1p, const float* __restrict__ root2,
                           const float* __restrict__ bias2, const float* __restrict__ aggr2,
                           const float* __restrict__ cnt2, float* __restrict__ x2) {
  int idx = blockIdx.x * blockDim.x + threadIdx.x;
  if (idx >= N1n * 64) return;
  int n = idx >> 6, o = idx & 63;
  float acc = bias2[o];
#pragma unroll
  for (int i = 0; i < 32; ++i) acc += x1p[n * 32 + i] * root2[i * 64 + o];
  acc += aggr2[idx] / fmaxf(cnt2[n], 1.0f);
  x2[idx] = eluf(acc);
}

// ---------------------------------------------------------------- pool 2 + global mean pool
__global__ void pool2(const float* __restrict__ x2, const int* __restrict__ batch1,
                      const int* __restrict__ cluster2, unsigned* __restrict__ x2p_enc,
                      int* __restrict__ batch2) {
  int n = blockIdx.x * blockDim.x + threadIdx.x;
  if (n >= N1n) return;
  int c = cluster2[n];
#pragma unroll
  for (int o = 0; o < 64; ++o) atomicMax(&x2p_enc[c * 64 + o], encf(x2[n * 64 + o]));
  atomicMax(&batch2[c], batch1[n]);
}

__global__ void gpool(const unsigned* __restrict__ x2p_enc, const int* __restrict__ batch2,
                      float* __restrict__ xg_sum, float* __restrict__ xg_cnt) {
  int n = blockIdx.x * blockDim.x + threadIdx.x;
  if (n >= N2n) return;
  int b = batch2[n];
  b = (b < 0) ? 0 : ((b >= Bb) ? Bb - 1 : b);
  atomicAdd(&xg_cnt[b], 1.0f);
#pragma unroll
  for (int o = 0; o < 64; ++o) {
    unsigned u = x2p_enc[n * 64 + o];
    float v = (u == ENC_NEGINF) ? 0.0f : decf(u);
    atomicAdd(&xg_sum[b * 64 + o], v);
  }
}

// ---------------------------------------------------------------- FC head + log_softmax
__global__ __launch_bounds__(32) void head_kernel(
    const float* __restrict__ xg_sum, const float* __restrict__ xg_cnt,
    const float* __restrict__ fc1w, const float* __restrict__ fc1b,
    const float* __restrict__ fc2w, const float* __restrict__ fc2b,
    float* __restrict__ out) {
  __shared__ float xg[64];
  __shared__ float h[128];
  __shared__ float lg[NCLSn];
  __shared__ float stats[2];
  int b = blockIdx.x;
  int lane = threadIdx.x;
  float cnt = fmaxf(xg_cnt[b], 1.0f);
  for (int i = lane; i < 64; i += 32) xg[i] = xg_sum[b * 64 + i] / cnt;
  __syncthreads();
  for (int u = lane; u < 128; u += 32) {
    float acc = fc1b[u];
    for (int i = 0; i < 64; ++i) acc += xg[i] * fc1w[i * 128 + u];
    h[u] = eluf(acc);
  }
  __syncthreads();
  if (lane < NCLSn) {
    float acc = fc2b[lane];
    for (int i = 0; i < 128; ++i) acc += h[i] * fc2w[i * NCLSn + lane];
    lg[lane] = acc;
  }
  __syncthreads();
  if (lane == 0) {
    float mx = lg[0];
    for (int c = 1; c < NCLSn; ++c) mx = fmaxf(mx, lg[c]);
    float s = 0.0f;
    for (int c = 0; c < NCLSn; ++c) s += expf(lg[c] - mx);
    stats[0] = mx;
    stats[1] = logf(s);
  }
  __syncthreads();
  if (lane < NCLSn) out[b * NCLSn + lane] = lg[lane] - stats[0] - stats[1];
}

// ---------------------------------------------------------------- host launcher
extern "C" void kernel_launch(void* const* d_in, const int* in_sizes, int n_in,
                              void* d_out, int out_size, void* d_ws, size_t ws_size,
                              hipStream_t stream) {
  (void)in_sizes; (void)n_in; (void)out_size; (void)ws_size;
  const float* x        = (const float*)d_in[0];
  const float* pos      = (const float*)d_in[1];
  const float* eattr    = (const float*)d_in[2];
  const int*   ei       = (const int*)d_in[3];
  const int*   batch    = (const int*)d_in[4];
  const int*   cluster1 = (const int*)d_in[5];
  const int*   ei2      = (const int*)d_in[6];
  const int*   cluster2 = (const int*)d_in[7];
  const float* w1_1  = (const float*)d_in[8];
  const float* b1_1  = (const float*)d_in[9];
  const float* w2_1  = (const float*)d_in[10];
  const float* b2_1  = (const float*)d_in[11];
  const float* root1 = (const float*)d_in[12];
  const float* bias1 = (const float*)d_in[13];
  const float* w1_2  = (const float*)d_in[14];
  const float* b1_2  = (const float*)d_in[15];
  const float* w2_2  = (const float*)d_in[16];
  const float* b2_2  = (const float*)d_in[17];
  const float* root2 = (const float*)d_in[18];
  const float* bias2 = (const float*)d_in[19];
  const float* fc1w  = (const float*)d_in[20];
  const float* fc1b  = (const float*)d_in[21];
  const float* fc2w  = (const float*)d_in[22];
  const float* fc2b  = (const float*)d_in[23];
  float* out = (float*)d_out;

  char* ws = (char*)d_ws;
  size_t off = 0;
  auto alloc = [&](size_t bytes) -> char* {
    char* p = ws + off;
    off = (off + bytes + 255) & ~size_t(255);
    return p;
  };
  float*    aggr1    = (float*)   alloc((size_t)Nn * 32 * 4);
  float*    cnt1     = (float*)   alloc((size_t)Nn * 4);
  float*    x1       = (float*)   alloc((size_t)Nn * 32 * 4);
  unsigned* x1p_enc  = (unsigned*)alloc((size_t)N1n * 32 * 4);
  float*    x1p      = (float*)   alloc((size_t)N1n * 32 * 4);
  float*    pos_sum  = (float*)   alloc((size_t)N1n * 2 * 4);
  float*    pos_cnt  = (float*)   alloc((size_t)N1n * 4);
  float*    pos1     = (float*)   alloc((size_t)N1n * 2 * 4);
  int*      batch1   = (int*)     alloc((size_t)N1n * 4);
  unsigned* m_enc    = (unsigned*)alloc(256);
  _Float16* w2h      = (_Float16*)alloc((size_t)26 * 2048 * 2);
  float*    aggr2    = (float*)   alloc((size_t)N1n * 64 * 4);
  float*    cnt2     = (float*)   alloc((size_t)N1n * 4);
  float*    x2       = (float*)   alloc((size_t)N1n * 64 * 4);
  unsigned* x2p_enc  = (unsigned*)alloc((size_t)N2n * 64 * 4);
  int*      batch2   = (int*)     alloc((size_t)N2n * 4);
  float*    xg_sum   = (float*)   alloc((size_t)Bb * 64 * 4);
  float*    xg_cnt   = (float*)   alloc((size_t)Bb * 4);

  // re-initialize accumulators every call (harness does not re-poison)
  hipMemsetAsync(aggr1,   0, (size_t)Nn * 32 * 4, stream);
  hipMemsetAsync(cnt1,    0, (size_t)Nn * 4, stream);
  hipMemsetAsync(pos_sum, 0, (size_t)N1n * 2 * 4, stream);
  hipMemsetAsync(pos_cnt, 0, (size_t)N1n * 4, stream);
  hipMemsetAsync(batch1,  0, (size_t)N1n * 4, stream);
  hipMemsetAsync(aggr2,   0, (size_t)N1n * 64 * 4, stream);
  hipMemsetAsync(cnt2,    0, (size_t)N1n * 4, stream);
  hipMemsetAsync(batch2,  0, (size_t)N2n * 4, stream);
  hipMemsetAsync(xg_sum,  0, (size_t)Bb * 64 * 4, stream);
  hipMemsetAsync(xg_cnt,  0, (size_t)Bb * 4, stream);
  fill_u32<<<(N1n * 32 + 255) / 256, 256, 0, stream>>>(x1p_enc, ENC_NEGINF, N1n * 32);
  fill_u32<<<(N2n * 64 + 255) / 256, 256, 0, stream>>>(x2p_enc, ENC_NEGINF, N2n * 64);
  fill_u32<<<1, 32, 0, stream>>>(m_enc, 0x80000000u, 1);  // enc(+0.0)
  prep_w2h<<<(26 * 2048 + 255) / 256, 256, 0, stream>>>(w2_2, b2_2, w2h);

  // conv1 + ELU
  conv1_edge<<<Ee / 256, 256, 0, stream>>>(x, eattr, ei, w1_1, b1_1, w2_1, b2_1, aggr1, cnt1);
  conv1_node<<<(Nn * 32 + 255) / 256, 256, 0, stream>>>(x, root1, bias1, aggr1, cnt1, x1);

  // graclus pool 1
  pool1<<<(Nn + 255) / 256, 256, 0, stream>>>(x1, pos, batch, cluster1,
                                              x1p_enc, pos_sum, pos_cnt, batch1);
  pos1_div<<<(N1n + 255) / 256, 256, 0, stream>>>(pos_sum, pos_cnt, pos1);
  x1p_decode<<<(N1n * 32 + 255) / 256, 256, 0, stream>>>(x1p_enc, x1p);
  cart_max<<<(E2n + 255) / 256, 256, 0, stream>>>(ei2, pos1, m_enc);

  // conv2 (fused edge MLP + einsum via WMMA) + node side + ELU
  conv2_edge_wmma<<<E2n / 128, 128, 0, stream>>>(ei2, pos1, m_enc, x1p,
                                                 w1_2, b1_2, w2h, aggr2, cnt2);
  conv2_node<<<(N1n * 64 + 255) / 256, 256, 0, stream>>>(x1p, root2, bias2, aggr2, cnt2, x2);

  // pool 2 + global mean pool + head
  pool2<<<(N1n + 255) / 256, 256, 0, stream>>>(x2, batch1, cluster2, x2p_enc, batch2);
  gpool<<<(N2n + 255) / 256, 256, 0, stream>>>(x2p_enc, batch2, xg_sum, xg_cnt);
  head_kernel<<<Bb, 32, 0, stream>>>(xg_sum, xg_cnt, fc1w, fc1b, fc2w, fc2b, out);
}
